// UV_Aggregator_31069793419778
// MI455X (gfx1250) — matrix operand order repaired and checked by reading, not compiled, not run on hardware
//
#include <hip/hip_runtime.h>
#include <hip/hip_bf16.h>

typedef __attribute__((ext_vector_type(16))) _Float16 v16h;
typedef __attribute__((ext_vector_type(8)))  _Float16 v8h;
typedef __attribute__((ext_vector_type(8)))  float    v8f;

#define NN 2048
#define LL 200
#define DD 64
#define WAVES 4
#define NCHUNK 13   // ceil(200/16)

#define WMMA_F16(A,B,C) __builtin_amdgcn_wmma_f32_16x16x32_f16(false,(A),false,(B),(short)0,(C),false,false)

static __device__ __forceinline__ v16h make16(v8h a, v8h b){
  v16h r;
#pragma unroll
  for (int i = 0; i < 8; ++i){ r[i] = a[i]; r[i+8] = b[i]; }
  return r;
}

// B fragment (32x16 f16): lane (n16,hi) holds column n16, halfs 0..7 -> K = k0+hi*8.., halfs 8..15 -> K = k0+16+hi*8..
static __device__ __forceinline__ v16h loadB16(const _Float16* w, int ldk, int n0, int k0, int lane){
  const int n16 = lane & 15, hi = lane >> 4;
  const _Float16* p = w + (n0 + n16) * ldk + k0 + hi*8;
  return make16(*(const v8h*)p, *(const v8h*)(p + 16));
}

// A fragment (16x32 f16) from LDS row-major [16][ldk]:
// lane (m16,hi): halfs 0..7 -> K = k0+hi*8.., halfs 8..15 -> K = k0+16+hi*8..
static __device__ __forceinline__ v16h loadA16(const _Float16* buf, int ldk, int k0, int lane){
  const int m16 = lane & 15, hi = lane >> 4;
  const _Float16* p = buf + m16*ldk + k0 + hi*8;
  return make16(*(const v8h*)p, *(const v8h*)(p + 16));
}

// A fragment gathered straight from a global f32 embedding table row
static __device__ __forceinline__ v16h gatherA16(const float* __restrict__ tab, int row, int k0, int lane){
  const int hi = lane >> 4;
  const float* p = tab + row*DD + k0 + hi*8;
  float4 f0 = *(const float4*)(p);
  float4 f1 = *(const float4*)(p + 4);
  float4 g0 = *(const float4*)(p + 16);
  float4 g1 = *(const float4*)(p + 20);
  v16h r;
  r[0]=(_Float16)f0.x; r[1]=(_Float16)f0.y; r[2]=(_Float16)f0.z; r[3]=(_Float16)f0.w;
  r[4]=(_Float16)f1.x; r[5]=(_Float16)f1.y; r[6]=(_Float16)f1.z; r[7]=(_Float16)f1.w;
  r[8]=(_Float16)g0.x; r[9]=(_Float16)g0.y; r[10]=(_Float16)g0.z; r[11]=(_Float16)g0.w;
  r[12]=(_Float16)g1.x; r[13]=(_Float16)g1.y; r[14]=(_Float16)g1.z; r[15]=(_Float16)g1.w;
  return r;
}

__global__ __launch_bounds__(WAVES*32, 1)
void uv_aggregator_kernel(const int* __restrict__ nodes,
                          const int* __restrict__ hist_uv,
                          const int* __restrict__ hist_r,
                          const float* __restrict__ u2e,
                          const float* __restrict__ v2e,
                          const float* __restrict__ r2e,
                          const float* __restrict__ w_r1_w, const float* __restrict__ w_r1_b,
                          const float* __restrict__ w_r2_w, const float* __restrict__ w_r2_b,
                          const float* __restrict__ mu_w,   const float* __restrict__ mu_b,
                          const float* __restrict__ att1_w, const float* __restrict__ att1_b,
                          const float* __restrict__ att2_w, const float* __restrict__ att2_b,
                          const float* __restrict__ att3_w, const float* __restrict__ att3_b,
                          float* __restrict__ out)
{
  // ---- f16 weight staging in LDS (CDNA5: 320KB/WGP) ----
  __shared__ _Float16 s_wr1[DD*128];
  __shared__ _Float16 s_mu [DD*128];
  __shared__ _Float16 s_at1[DD*128];
  __shared__ _Float16 s_wr2[DD*DD];
  __shared__ _Float16 s_at2[DD*DD];
  __shared__ float    s_bias[5*DD];
  __shared__ float    s_a3w[DD];
  __shared__ float    s_a3b;
  // per-wave staging
  __shared__ _Float16 s_oq [WAVES][16*128];  // [o ; q] rows (gate/att1 input)
  __shared__ _Float16 s_tmp[WAVES][16*DD];   // layer1 / att1 outputs
  __shared__ float    s_q  [WAVES][DD];
  __shared__ float    s_lg [WAVES][16];

  const int tid = threadIdx.x;
  for (int i = tid; i < DD*128; i += WAVES*32){
    s_wr1[i] = (_Float16)w_r1_w[i];
    s_mu [i] = (_Float16)mu_w[i];
    s_at1[i] = (_Float16)att1_w[i];
  }
  for (int i = tid; i < DD*DD; i += WAVES*32){
    s_wr2[i] = (_Float16)w_r2_w[i];
    s_at2[i] = (_Float16)att2_w[i];
  }
  for (int i = tid; i < DD; i += WAVES*32){
    s_bias[0*DD+i] = w_r1_b[i];
    s_bias[1*DD+i] = w_r2_b[i];
    s_bias[2*DD+i] = mu_b[i];
    s_bias[3*DD+i] = att1_b[i];
    s_bias[4*DD+i] = att2_b[i];
    s_a3w[i]       = att3_w[i];
  }
  if (tid == 0) s_a3b = att3_b[0];
  __syncthreads();

  const int wave = tid >> 5;
  const int lane = tid & 31;
  const int m16  = lane & 15;   // row (A/C) or column (B/C) index within tile
  const int hi   = lane >> 4;
  const int n    = blockIdx.x * WAVES + wave;

  // q = v2e[nodes[n]] : shared by all 200 rows of this wave
  const float* qp = v2e + (long)nodes[n] * DD;
  float q0 = qp[lane*2 + 0], q1 = qp[lane*2 + 1];
  s_q[wave][lane*2 + 0] = q0;
  s_q[wave][lane*2 + 1] = q1;
  {
    _Float16 q0h = (_Float16)q0, q1h = (_Float16)q1;
#pragma unroll
    for (int m = 0; m < 16; ++m){
      s_oq[wave][m*128 + DD + lane*2 + 0] = q0h;
      s_oq[wave][m*128 + DD + lane*2 + 1] = q1h;
    }
  }

  // online-softmax running state (uniform across lanes except acc)
  float run_max = -1e30f, run_sum = 0.f;
  float acc[4] = {0.f, 0.f, 0.f, 0.f};

  for (int ch = 0; ch < NCHUNK; ++ch){
    const int l  = ch*16 + m16;
    const int lc = (l < LL) ? l : (LL - 1);
    const int iu = hist_uv[n*LL + lc];
    const int ir = hist_r [n*LL + lc];

    // ---- A fragments for layer1 input [e_uv ; e_r] (K = 128) ----
    v16h A1[4];
    A1[0] = gatherA16(u2e, iu, 0 , lane);
    A1[1] = gatherA16(u2e, iu, 32, lane);
    A1[2] = gatherA16(r2e, ir, 0 , lane);
    A1[3] = gatherA16(r2e, ir, 32, lane);

    // ---- layer1: x = relu([e_uv;e_r] W1^T + b1) ----
#pragma unroll
    for (int nt = 0; nt < 4; ++nt){
      v8f C = {};
#pragma unroll
      for (int k = 0; k < 4; ++k)
        C = WMMA_F16(A1[k], loadB16(s_wr1, 128, nt*16, k*32, lane), C);
      float b = s_bias[0*DD + nt*16 + m16];
#pragma unroll
      for (int r = 0; r < 8; ++r){
        float x = C[r] + b; x = x > 0.f ? x : 0.f;
        s_tmp[wave][(hi*8 + r)*DD + nt*16 + m16] = (_Float16)x;
      }
    }

    // ---- layer2: o = relu(x W2^T + b2) ----
    v16h A2[2];
    A2[0] = loadA16(s_tmp[wave], DD, 0 , lane);
    A2[1] = loadA16(s_tmp[wave], DD, 32, lane);
    v8f Co[4];
#pragma unroll
    for (int nt = 0; nt < 4; ++nt){
      v8f C = {};
      C = WMMA_F16(A2[0], loadB16(s_wr2, DD, nt*16, 0 , lane), C);
      C = WMMA_F16(A2[1], loadB16(s_wr2, DD, nt*16, 32, lane), C);
      float b = s_bias[1*DD + nt*16 + m16];
#pragma unroll
      for (int r = 0; r < 8; ++r){
        float x = C[r] + b; x = x > 0.f ? x : 0.f;
        Co[nt][r] = x;
        s_oq[wave][(hi*8 + r)*128 + nt*16 + m16] = (_Float16)x;
      }
    }

    // ---- gate: g = sigmoid([o;q] Mu^T + mb); o = g*o + (1-g)*q ----
    v16h Am[4];
#pragma unroll
    for (int k = 0; k < 4; ++k) Am[k] = loadA16(s_oq[wave], 128, k*32, lane);
#pragma unroll
    for (int nt = 0; nt < 4; ++nt){
      v8f C = {};
#pragma unroll
      for (int k = 0; k < 4; ++k)
        C = WMMA_F16(Am[k], loadB16(s_mu, 128, nt*16, k*32, lane), C);
      float b  = s_bias[2*DD + nt*16 + m16];
      float qv = s_q[wave][nt*16 + m16];
#pragma unroll
      for (int r = 0; r < 8; ++r){
        float g  = 1.f / (1.f + __expf(-(C[r] + b)));
        float og = g * Co[nt][r] + (1.f - g) * qv;
        Co[nt][r] = og;
        s_oq[wave][(hi*8 + r)*128 + nt*16 + m16] = (_Float16)og;
      }
    }

    // ---- att1: a = relu([o;q] A1^T + b) ----
    v16h Aa[4];
#pragma unroll
    for (int k = 0; k < 4; ++k) Aa[k] = loadA16(s_oq[wave], 128, k*32, lane);
#pragma unroll
    for (int nt = 0; nt < 4; ++nt){
      v8f C = {};
#pragma unroll
      for (int k = 0; k < 4; ++k)
        C = WMMA_F16(Aa[k], loadB16(s_at1, 128, nt*16, k*32, lane), C);
      float b = s_bias[3*DD + nt*16 + m16];
#pragma unroll
      for (int r = 0; r < 8; ++r){
        float x = C[r] + b; x = x > 0.f ? x : 0.f;
        s_tmp[wave][(hi*8 + r)*DD + nt*16 + m16] = (_Float16)x;
      }
    }

    // ---- att2 + att3 dot -> per-row logits ----
    v16h Ab[2];
    Ab[0] = loadA16(s_tmp[wave], DD, 0 , lane);
    Ab[1] = loadA16(s_tmp[wave], DD, 32, lane);
    float part[8] = {0.f,0.f,0.f,0.f,0.f,0.f,0.f,0.f};
#pragma unroll
    for (int nt = 0; nt < 4; ++nt){
      v8f C = {};
      C = WMMA_F16(Ab[0], loadB16(s_at2, DD, nt*16, 0 , lane), C);
      C = WMMA_F16(Ab[1], loadB16(s_at2, DD, nt*16, 32, lane), C);
      float b  = s_bias[4*DD + nt*16 + m16];
      float w3 = s_a3w[nt*16 + m16];
#pragma unroll
      for (int r = 0; r < 8; ++r){
        float x = C[r] + b; x = x > 0.f ? x : 0.f;
        part[r] += w3 * x;
      }
    }
    // reduce the 64-wide dot across the 16 lanes of each half-wave
#pragma unroll
    for (int off = 1; off < 16; off <<= 1){
#pragma unroll
      for (int r = 0; r < 8; ++r)
        part[r] += __shfl_xor(part[r], off, 32);
    }
    if (m16 == 0){
#pragma unroll
      for (int r = 0; r < 8; ++r){
        int m = hi*8 + r;
        s_lg[wave][m] = (ch*16 + m < LL) ? (part[r] + s_a3b) : -1e30f;
      }
    }

    // ---- online softmax + weighted accumulation (flash-style) ----
    float lg[16];
    float cmax = -1e30f;
#pragma unroll
    for (int i = 0; i < 16; ++i){ lg[i] = s_lg[wave][i]; cmax = fmaxf(cmax, lg[i]); }
    float newm  = fmaxf(run_max, cmax);
    float scale = __expf(run_max - newm);
    float ssum  = 0.f;
#pragma unroll
    for (int i = 0; i < 16; ++i) ssum += __expf(lg[i] - newm);
    run_sum = run_sum * scale + ssum;
    run_max = newm;
#pragma unroll
    for (int nt = 0; nt < 4; ++nt) acc[nt] *= scale;
#pragma unroll
    for (int r = 0; r < 8; ++r){
      float w = __expf(lg[hi*8 + r] - newm);
#pragma unroll
      for (int nt = 0; nt < 4; ++nt) acc[nt] += w * Co[nt][r];
    }
  }

  // ---- epilogue: fold the two half-wave row groups, normalize, store ----
#pragma unroll
  for (int nt = 0; nt < 4; ++nt){
    float tot = acc[nt] + __shfl_xor(acc[nt], 16, 32);
    if (hi == 0)
      out[n*DD + nt*16 + m16] = tot / run_sum;
  }
}

extern "C" void kernel_launch(void* const* d_in, const int* in_sizes, int n_in,
                              void* d_out, int out_size, void* d_ws, size_t ws_size,
                              hipStream_t stream) {
  const int*   nodes  = (const int*)  d_in[0];
  const int*   huv    = (const int*)  d_in[1];
  const int*   hr     = (const int*)  d_in[2];
  const float* u2e    = (const float*)d_in[3];
  const float* v2e    = (const float*)d_in[4];
  const float* r2e    = (const float*)d_in[5];
  const float* w_r1_w = (const float*)d_in[6];
  const float* w_r1_b = (const float*)d_in[7];
  const float* w_r2_w = (const float*)d_in[8];
  const float* w_r2_b = (const float*)d_in[9];
  const float* mu_w   = (const float*)d_in[10];
  const float* mu_b   = (const float*)d_in[11];
  const float* att1_w = (const float*)d_in[12];
  const float* att1_b = (const float*)d_in[13];
  const float* att2_w = (const float*)d_in[14];
  const float* att2_b = (const float*)d_in[15];
  const float* att3_w = (const float*)d_in[16];
  const float* att3_b = (const float*)d_in[17];
  float* out = (float*)d_out;

  dim3 grid(NN / WAVES);
  dim3 block(WAVES * 32);
  hipLaunchKernelGGL(uv_aggregator_kernel, grid, block, 0, stream,
                     nodes, huv, hr, u2e, v2e, r2e,
                     w_r1_w, w_r1_b, w_r2_w, w_r2_b, mu_w, mu_b,
                     att1_w, att1_b, att2_w, att2_b, att3_w, att3_b, out);
}
